// RQVAETokenizer_84258668413392
// MI455X (gfx1250) — matrix operand order repaired and checked by reading, not compiled, not run on hardware
//
#include <hip/hip_runtime.h>

// ---------------------------------------------------------------------------
// RQ-VAE forward for MI455X (gfx1250): bf16 WMMA GEMMs + VALU residual-VQ.
//   B=65536, D_IN=768, D_HID=256, D_LAT=64, 3 levels x 256 codes.
// Weights are converted+transposed once to bf16 [N][K] (tiny, L2-resident),
// so BOTH GEMM operand tiles stream via GLOBAL_LOAD_ASYNC_TO_LDS_B128 into
// double-buffered LDS, paced with s_wait_asynccnt. The bf16-GEMM hot loop is
// pure: async issue -> wait -> barrier -> ds_load_b128 -> v_wmma (8/stage).
// ---------------------------------------------------------------------------

typedef __attribute__((ext_vector_type(16))) __bf16 v16bf;
typedef __attribute__((ext_vector_type(8)))  float  v8f;

union FragBF { unsigned int u[8]; v16bf v; };

__device__ __forceinline__ unsigned short f32_to_bf16(float f) {
  unsigned int u = __float_as_uint(f);
  unsigned int r = (u + 0x7FFFu + ((u >> 16) & 1u)) >> 16;   // round-nearest-even
  return (unsigned short)r;
}
__device__ __forceinline__ unsigned int pack_bf16x2(float lo, float hi) {
  return (unsigned int)f32_to_bf16(lo) | ((unsigned int)f32_to_bf16(hi) << 16);
}

// Low 32 bits of a generic pointer to LDS are the LDS byte offset
// (local->generic addrspacecast is {aperture_hi, lds_offset}; ISA 10.2).
__device__ __forceinline__ unsigned lds_off(const void* p) {
  return (unsigned)(unsigned long long)p;
}
// Async DMA: 16 bytes global -> LDS, tracked on ASYNCcnt (ISA 15.18.3 op 98).
__device__ __forceinline__ void async_copy_b128(unsigned lds_byte_off, const void* gaddr) {
  asm volatile("global_load_async_to_lds_b128 %0, %1, off"
               :: "v"(lds_byte_off), "v"(gaddr) : "memory");
}
template<int N>
__device__ __forceinline__ void wait_async() {
  if      (N == 6) asm volatile("s_wait_asynccnt 0x6" ::: "memory");
  else if (N == 2) asm volatile("s_wait_asynccnt 0x2" ::: "memory");
  else             asm volatile("s_wait_asynccnt 0x0" ::: "memory");
}

// ---------------------------------------------------------------------------
// One-shot weight prep: W[K,N] f32 -> Wt[N,K] bf16 (coalesced writes).
// ---------------------------------------------------------------------------
__global__ __launch_bounds__(256) void convert_transpose_w(
    const float* __restrict__ W, unsigned short* __restrict__ Wt, int K, int N)
{
  int idx = blockIdx.x * 256 + threadIdx.x;     // over N*K
  int n = idx / K, k = idx - n * K;
  Wt[idx] = f32_to_bf16(W[(size_t)k * N + n]);
}

// ---------------------------------------------------------------------------
// Out[M,N] = act(A[M,K] @ Wt[N,K]^T + bias[N])
// Block tile 128x64, K-stage 64, 256 threads = 8 waves, wave tile 16x64.
// sA: rows of 32 dwords, row stride 36 (16B-aligned rows, conflict-free).
// sB: column-major, K-pairs contiguous per column, column stride 36 dwords.
// Fragments of both operands read as contiguous 8-dword runs -> ds_load_b128.
// ---------------------------------------------------------------------------
template<bool A_IS_F32, bool OUT_F32, bool RELU>
__global__ __launch_bounds__(256) void gemm_wmma(
    const void* __restrict__ Aptr, const unsigned short* __restrict__ Wt,
    const float* __restrict__ bias, void* __restrict__ Out,
    int M, int K, int N)
{
  __shared__ unsigned int sA[2][128 * 36];  // 2 x (128 rows x 32 dwords + pad)
  __shared__ unsigned int sB[2][64 * 36];   // 2 x ([n][k/2] packed pairs, K=64)

  const int m0   = blockIdx.x * 128;
  const int n0   = blockIdx.y * 64;
  const int t    = threadIdx.x;
  const int lane = t & 31;
  const int half = lane >> 4;
  const int ln16 = lane & 15;
  const int mw   = (t >> 5) * 16;           // wave's row offset in block tile
  const int S    = K >> 6;                  // number of 64-wide K stages

  v8f acc[4] = {};
  float4 apf[8];                            // A tile prefetch (f32 path only)

  // ---------------- prologue: stage 0 in flight ----------------
  #pragma unroll
  for (int it = 0; it < 2; ++it) {          // B tile: 64 cols x 64 k bf16
    int c = it * 256 + t, col = c >> 3, kc = (c & 7) * 8;
    async_copy_b128(lds_off(&sB[0][col * 36 + (kc >> 1)]),
                    Wt + (size_t)(n0 + col) * K + kc);
  }
  if (A_IS_F32) {
    #pragma unroll
    for (int it = 0; it < 8; ++it) {
      int c = it * 256 + t, row = c >> 4, col = (c & 15) * 4;
      apf[it] = *(const float4*)((const float*)Aptr + (size_t)(m0 + row) * K + col);
    }
  } else {
    #pragma unroll
    for (int it = 0; it < 4; ++it) {
      int c = it * 256 + t, row = c >> 3, kc = (c & 7) * 8;
      const void* g = (const unsigned short*)Aptr + (size_t)(m0 + row) * K + kc;
      async_copy_b128(lds_off(&sA[0][row * 36 + (kc >> 1)]), g);
    }
  }

  for (int s = 0; s < S; ++s) {
    const int p  = s & 1;
    const int kt = s * 64;

    if (A_IS_F32) {                         // convert prefetched A regs -> LDS
      #pragma unroll
      for (int it = 0; it < 8; ++it) {
        int c = it * 256 + t, row = c >> 4, col = (c & 15) * 4;
        sA[p][row * 36 + (col >> 1) + 0] = pack_bf16x2(apf[it].x, apf[it].y);
        sA[p][row * 36 + (col >> 1) + 1] = pack_bf16x2(apf[it].z, apf[it].w);
      }
    }

    // ---- issue stage s+1 DMAs (overlap with this stage's compute) ----
    const bool more = (s + 1 < S);
    if (more) {
      if (A_IS_F32) {
        #pragma unroll
        for (int it = 0; it < 8; ++it) {
          int c = it * 256 + t, row = c >> 4, col = (c & 15) * 4;
          apf[it] = *(const float4*)((const float*)Aptr +
                                     (size_t)(m0 + row) * K + kt + 64 + col);
        }
      } else {
        #pragma unroll
        for (int it = 0; it < 4; ++it) {
          int c = it * 256 + t, row = c >> 3, kc = (c & 7) * 8;
          const void* g = (const unsigned short*)Aptr +
                          (size_t)(m0 + row) * K + kt + 64 + kc;
          async_copy_b128(lds_off(&sA[p ^ 1][row * 36 + (kc >> 1)]), g);
        }
      }
      #pragma unroll
      for (int it = 0; it < 2; ++it) {
        int c = it * 256 + t, col = c >> 3, kc = (c & 7) * 8;
        async_copy_b128(lds_off(&sB[p ^ 1][col * 36 + (kc >> 1)]),
                        Wt + (size_t)(n0 + col) * K + kt + 64 + kc);
      }
    }
    // stage s's DMAs complete; stage s+1's (6 or 2) may stay in flight
    if (more) { if (A_IS_F32) wait_async<2>(); else wait_async<6>(); }
    else      wait_async<0>();
    __syncthreads();

    // ---- compute: 2 K-substeps x 4 N-tiles = 8 WMMA ----
    #pragma unroll
    for (int ks = 0; ks < 2; ++ks) {
      FragBF a;
      const int row = mw + ln16;
      #pragma unroll
      for (int v = 0; v < 8; ++v) {
        int kd = ks * 16 + half * 8 + v;  // dword (K/2) idx
        a.u[v] = sA[p][row * 36 + kd];
      }
      #pragma unroll
      for (int nt = 0; nt < 4; ++nt) {
        FragBF bfr;
        const int base = (nt * 16 + ln16) * 36 + ks * 16 + half * 8;
        #pragma unroll
        for (int v = 0; v < 8; ++v)          // 8 contiguous dwords -> 2x b128
          bfr.u[v] = sB[p][base + v];
        acc[nt] = __builtin_amdgcn_wmma_f32_16x16x32_bf16(
            false, a.v, false, bfr.v, (short)0, acc[nt], false, false);
      }
    }
    __syncthreads();
  }

  // ---- epilogue: bias (+ReLU), store ----
  #pragma unroll
  for (int nt = 0; nt < 4; ++nt) {
    int col  = n0 + nt * 16 + ln16;
    float bv = bias[col];
    #pragma unroll
    for (int v = 0; v < 8; ++v) {
      int row   = m0 + mw + v + 8 * half;
      float val = acc[nt][v] + bv;
      if (RELU) val = fmaxf(val, 0.0f);
      if (OUT_F32) ((float*)Out)[(size_t)row * N + col] = val;
      else ((unsigned short*)Out)[(size_t)row * N + col] = f32_to_bf16(val);
    }
  }
}

// ---------------------------------------------------------------------------
// Residual VQ: one thread per row, codebook chunks (128x64 f32) in LDS.
// dist(k) = ||r||^2 + (||E_k||^2 - 2 r.E_k); argmin; gather; residual update.
// closs = sum_l (1+beta) * mean((r-e)^2);  quantized(fwd) = sum_l e.
// ---------------------------------------------------------------------------
__global__ __launch_bounds__(256) void rvq_kernel(
    const float* __restrict__ z, const float* __restrict__ codebooks,
    unsigned short* __restrict__ q_bf16, float* __restrict__ codes_out,
    float* __restrict__ loss_out)
{
  __shared__ float sE[128 * 64];
  __shared__ float sE2[128];
  __shared__ float sRed[256];

  const int t = threadIdx.x;
  const int b = blockIdx.x * 256 + t;

  float r[64], qs[64];
  #pragma unroll
  for (int d = 0; d < 64; ++d) { r[d] = z[(size_t)b * 64 + d]; qs[d] = 0.0f; }

  float closs = 0.0f;
  for (int l = 0; l < 3; ++l) {
    float r2 = 0.0f;
    #pragma unroll
    for (int d = 0; d < 64; ++d) r2 = fmaf(r[d], r[d], r2);

    float best = 3.4e38f;
    int bestk  = 0;
    for (int ch = 0; ch < 2; ++ch) {
      __syncthreads();                       // previous chunk scan done
      const float* Ech = codebooks + ((size_t)l * 256 + (size_t)ch * 128) * 64;
      for (int i = t; i < 2048; i += 256)    // 128x64 floats as float4
        ((float4*)sE)[i] = ((const float4*)Ech)[i];
      __syncthreads();
      if (t < 128) {
        float s = 0.0f;
        #pragma unroll 8
        for (int d = 0; d < 64; ++d) { float e = sE[t * 64 + d]; s = fmaf(e, e, s); }
        sE2[t] = s;
      }
      __syncthreads();
      for (int k = 0; k < 128; ++k) {
        float dot = 0.0f;
        #pragma unroll
        for (int d = 0; d < 64; ++d) dot = fmaf(r[d], sE[k * 64 + d], dot);
        float score = sE2[k] - 2.0f * dot;
        if (score < best) { best = score; bestk = ch * 128 + k; }
      }
    }
    float dist = r2 + best;
    if (dist < 0.0f) dist = 0.0f;
    closs += dist;
    // gather winning code from the (L2-resident) codebook
    const float* e = codebooks + ((size_t)l * 256 + bestk) * 64;
    #pragma unroll
    for (int d = 0; d < 64; ++d) { float ev = e[d]; qs[d] += ev; r[d] -= ev; }
    codes_out[(size_t)b * 3 + l] = (float)bestk;
  }

  #pragma unroll
  for (int d = 0; d < 64; ++d)
    q_bf16[(size_t)b * 64 + d] = f32_to_bf16(qs[d]);

  // block-reduce commitment loss, scaled: (1+beta)/(B*D_LAT)
  sRed[t] = closs * (1.25f / (65536.0f * 64.0f));
  __syncthreads();
  for (int s = 128; s > 0; s >>= 1) {
    if (t < s) sRed[t] += sRed[t + s];
    __syncthreads();
  }
  if (t == 0) atomicAdd(loss_out, sRed[0]);
}

__global__ void zero_scalar(float* p) { *p = 0.0f; }

// ---------------------------------------------------------------------------
extern "C" void kernel_launch(void* const* d_in, const int* in_sizes, int n_in,
                              void* d_out, int out_size, void* d_ws, size_t ws_size,
                              hipStream_t stream) {
  (void)in_sizes; (void)n_in; (void)out_size; (void)ws_size;

  const float* x    = (const float*)d_in[0];
  const float* ew0  = (const float*)d_in[1];
  const float* eb0  = (const float*)d_in[2];
  const float* ew1  = (const float*)d_in[3];
  const float* eb1  = (const float*)d_in[4];
  const float* ew2  = (const float*)d_in[5];
  const float* eb2  = (const float*)d_in[6];
  const float* dw0  = (const float*)d_in[7];
  const float* db0  = (const float*)d_in[8];
  const float* dw1  = (const float*)d_in[9];
  const float* db1  = (const float*)d_in[10];
  const float* dw2  = (const float*)d_in[11];
  const float* db2  = (const float*)d_in[12];
  const float* cbk  = (const float*)d_in[13];

  const int B = 65536;

  // workspace carve-out (~93 MB)
  char* ws = (char*)d_ws;
  unsigned short* hA = (unsigned short*)(ws);                          // B*256 bf16
  unsigned short* hB = (unsigned short*)(ws + (size_t)B * 256 * 2);    // B*256 bf16
  float*          zb = (float*)         (ws + (size_t)B * 256 * 4);    // B*64  f32
  unsigned short* qb = (unsigned short*)(ws + (size_t)B * 256 * 4
                                            + (size_t)B * 64 * 4);     // B*64  bf16
  unsigned short* wts = (unsigned short*)(ws + (size_t)B * 256 * 4
                                             + (size_t)B * 64 * 6);
  unsigned short* tew0 = wts;             // [256][768]
  unsigned short* tew1 = tew0 + 196608;   // [256][256]
  unsigned short* tew2 = tew1 + 65536;    // [ 64][256]
  unsigned short* tdw0 = tew2 + 16384;    // [256][ 64]
  unsigned short* tdw1 = tdw0 + 16384;    // [256][256]
  unsigned short* tdw2 = tdw1 + 65536;    // [768][256]

  float* recon = (float*)d_out;                 // B*768
  float* loss  = recon + (size_t)B * 768;       // 1
  float* codes = loss + 1;                      // B*3 (float-encoded indices)

  dim3 blk(256);
  const int MB = B / 128;

  // one-shot weight convert+transpose: W[K,N] f32 -> Wt[N,K] bf16
  convert_transpose_w<<<dim3(196608 / 256), blk, 0, stream>>>(ew0, tew0, 768, 256);
  convert_transpose_w<<<dim3( 65536 / 256), blk, 0, stream>>>(ew1, tew1, 256, 256);
  convert_transpose_w<<<dim3( 16384 / 256), blk, 0, stream>>>(ew2, tew2, 256,  64);
  convert_transpose_w<<<dim3( 16384 / 256), blk, 0, stream>>>(dw0, tdw0,  64, 256);
  convert_transpose_w<<<dim3( 65536 / 256), blk, 0, stream>>>(dw1, tdw1, 256, 256);
  convert_transpose_w<<<dim3(196608 / 256), blk, 0, stream>>>(dw2, tdw2, 256, 768);
  zero_scalar<<<1, 1, 0, stream>>>(loss);

  // encoder
  gemm_wmma<true,  false, true ><<<dim3(MB,  4), blk, 0, stream>>>(x,  tew0, eb0, hA, B, 768, 256);
  gemm_wmma<false, false, true ><<<dim3(MB,  4), blk, 0, stream>>>(hA, tew1, eb1, hB, B, 256, 256);
  gemm_wmma<false, true,  false><<<dim3(MB,  1), blk, 0, stream>>>(hB, tew2, eb2, zb, B, 256, 64);
  // residual VQ
  rvq_kernel<<<dim3(B / 256), blk, 0, stream>>>(zb, cbk, qb, codes, loss);
  // decoder
  gemm_wmma<false, false, true ><<<dim3(MB,  4), blk, 0, stream>>>(qb, tdw0, db0, hA, B,  64, 256);
  gemm_wmma<false, false, true ><<<dim3(MB,  4), blk, 0, stream>>>(hA, tdw1, db1, hB, B, 256, 256);
  gemm_wmma<false, true,  false><<<dim3(MB, 12), blk, 0, stream>>>(hB, tdw2, db2, recon, B, 256, 768);
}
